// ProtGVPModel_8194797600812
// MI455X (gfx1250) — matrix operand order repaired and checked by reading, not compiled, not run on hardware
//
#include <hip/hip_runtime.h>
#include <math.h>

// ---------------------------------------------------------------------------
// GVP-GNN forward for MI455X (gfx1250, wave32).
// Heavy GEMMs: v_wmma_f32_16x16x32_bf16 (f32 accumulate). Weights are staged
// in LDS once per workgroup (row stride padded to K+8 -> conflict-free
// ds_load_b128 fragments), each wave owns 16x128 output strips via grid-stride
// so the A fragment is loaded once and feeds 8 independent WMMAs per k-step.
// Small vector-channel matmuls / layernorms / gates are scalar VALU.
// Mean aggregation over random edges = float atomics.
// ---------------------------------------------------------------------------

#define NN 10240      // nodes
#define NE 163840     // edges
#define EPSV 1e-8f

typedef __bf16 bf16_t;
typedef __attribute__((ext_vector_type(16))) __bf16 v16bf;
typedef __attribute__((ext_vector_type(8)))  __bf16 v8bf;
typedef __attribute__((ext_vector_type(8)))  float  v8f;

__device__ __forceinline__ unsigned short f2bf(float f) {
    union { float f; unsigned u; } x; x.f = f;
    unsigned r = (x.u + 0x7FFFu + ((x.u >> 16) & 1u)) >> 16;
    return (unsigned short)r;
}
__device__ __forceinline__ float bf2f(unsigned short b) {
    union { unsigned u; float f; } x; x.u = ((unsigned)b) << 16;
    return x.f;
}
__device__ __forceinline__ float sigm(float x) { return 1.f / (1.f + __expf(-x)); }

__device__ __forceinline__ void ln_inplace(float* x, int d, const float* g, const float* b) {
    float mu = 0.f;
    for (int i = 0; i < d; ++i) mu += x[i];
    mu /= (float)d;
    float var = 0.f;
    for (int i = 0; i < d; ++i) { float t = x[i] - mu; var += t * t; }
    var /= (float)d;
    float inv = rsqrtf(var + 1e-5f);
    for (int i = 0; i < d; ++i) x[i] = (x[i] - mu) * inv * g[i] + b[i];
}

// ---------------------------------------------------------------------------
// bf16 WMMA GEMM:  C(MxN,f32) = A(MxK,bf16 rowmajor) * B(KxN) + bias
// BT is the transposed weight, N x K (bf16), K = Kpad (multiple of 32).
// Block stages BT in LDS (row stride K+8: stride in dwords mod 64 has gcd 4
// with odd quotient -> the 16 fragment lanes hit disjoint bank groups), then
// its waves grid-stride over 16(M) x 128(N) output strips. N % 128 == 0 for
// every GEMM here (N = 128 or 512). Fragment layouts per CDNA5 ISA 7.12.2:
//   A 16-bit 16x32: lane<16 -> K{0..7,16..23}, lane>=16 -> K{8..15,24..31}
//   B 16-bit 32x16: lane<16 -> K{0..7,16..23}, lane>=16 -> K{8..15,24..31}
// ---------------------------------------------------------------------------
__global__ void gemm_bf16_k(const unsigned short* __restrict__ A, int lda,
                            const unsigned short* __restrict__ BT,
                            const float* __restrict__ bias,
                            float* __restrict__ C, int M, int N, int K, int act)
{
    extern __shared__ unsigned short Bs[];      // N x (K+8) bf16
    const int Klds = K + 8;
    const int tid  = threadIdx.x;

    // ---- cooperative global -> LDS copy of BT, 16B chunks ----
    const int rowChunks = K >> 3;               // 8-element chunks per row
    const int chunks    = N * rowChunks;
    for (int cI = tid; cI < chunks; cI += blockDim.x) {
        int nrow = cI / rowChunks;
        int off  = (cI - nrow * rowChunks) << 3;
        *(v8bf*)&Bs[(size_t)nrow * Klds + off] =
            *(const v8bf*)((const bf16_t*)BT + (size_t)nrow * K + off);
    }
    __syncthreads();

    const int lane = tid & 31;
    const int hi   = (lane >> 4) & 1;
    const int r    = lane & 15;
    const int stripsN = N >> 7;
    const int strips  = (M >> 4) * stripsN;
    const int wavesPerBlock = blockDim.x >> 5;
    const int gwave  = blockIdx.x * wavesPerBlock + (tid >> 5);
    const int nwaves = gridDim.x * wavesPerBlock;

    for (int s = gwave; s < strips; s += nwaves) {   // wave-uniform loop
        int tm = s / stripsN;
        int tn = s - tm * stripsN;
        const bf16_t* Ar = (const bf16_t*)A + (size_t)(tm * 16 + r) * lda + hi * 8;
        const unsigned short* Bl = &Bs[(size_t)(tn * 128 + r) * Klds + hi * 8];
        v8f acc[8] = {};
        for (int k = 0; k < K; k += 32) {
            v8bf a0 = *(const v8bf*)(Ar + k);
            v8bf a1 = *(const v8bf*)(Ar + k + 16);
            v16bf a;
#pragma unroll
            for (int i = 0; i < 8; ++i) { a[i] = a0[i]; a[i + 8] = a1[i]; }
#pragma unroll
            for (int t = 0; t < 8; ++t) {       // 8 independent WMMAs share A
                const unsigned short* bp = Bl + (size_t)(t * 16) * Klds + k;
                v8bf b0 = *(const v8bf*)bp;
                v8bf b1 = *(const v8bf*)(bp + 16);
                v16bf b;
#pragma unroll
                for (int i = 0; i < 8; ++i) { b[i] = b0[i]; b[i + 8] = b1[i]; }
                acc[t] = __builtin_amdgcn_wmma_f32_16x16x32_bf16(false, a, false, b,
                                                                 (short)0, acc[t], false, false);
            }
        }
        int rbase = tm * 16 + hi * 8;
#pragma unroll
        for (int t = 0; t < 8; ++t) {
            int col = tn * 128 + t * 16 + r;
            float bv = bias[col];
#pragma unroll
            for (int i = 0; i < 8; ++i) {
                float v = acc[t][i] + bv;
                if (act) v = fmaxf(v, 0.f);
                C[(size_t)(rbase + i) * N + col] = v;
            }
        }
    }
}

// fp32 (K x N) weight -> bf16 transposed (N x Kpad), zero padded K rows
__global__ void wconv_k(const float* __restrict__ W, unsigned short* __restrict__ WT,
                        int K, int N, int Kpad)
{
    int t = blockIdx.x * blockDim.x + threadIdx.x;
    if (t >= N * Kpad) return;
    int n = t / Kpad, k = t - n * Kpad;
    WT[t] = (k < K) ? f2bf(W[(size_t)k * N + n]) : (unsigned short)0;
}

__global__ void zero_k(float* __restrict__ p, int n)
{
    int i = blockIdx.x * blockDim.x + threadIdx.x;
    if (i < n) p[i] = 0.f;
}

// ---------------------------------------------------------------------------
// W_v embedding: LN(6) + GVP(6,3 -> 128,16), no activations
// ---------------------------------------------------------------------------
__global__ void embed_nodes_k(const float* __restrict__ s_in, const float* __restrict__ v_in,
                              const float* __restrict__ g, const float* __restrict__ be,
                              const float* __restrict__ wh,   /* 3x16 */
                              const float* __restrict__ ws_w, /* 22x128 */
                              const float* __restrict__ ws_b,
                              const float* __restrict__ wv,   /* 16x16 */
                              float* __restrict__ s_out, float* __restrict__ v_out)
{
    int nI = blockIdx.x * blockDim.x + threadIdx.x;
    if (nI >= NN) return;
    float s[6];
    for (int i = 0; i < 6; ++i) s[i] = s_in[nI * 6 + i];
    ln_inplace(s, 6, g, be);
    float v[3][3]; float q = 0.f;
    for (int i = 0; i < 3; ++i) {
        float ss = 0.f;
        for (int c = 0; c < 3; ++c) { v[i][c] = v_in[nI * 9 + i * 3 + c]; ss += v[i][c] * v[i][c]; }
        q += fmaxf(ss, EPSV);
    }
    float rms = sqrtf(q / 3.f);
    for (int i = 0; i < 3; ++i) for (int c = 0; c < 3; ++c) v[i][c] /= rms;
    float sv[22];
    for (int i = 0; i < 6; ++i) sv[i] = s[i];
    float vh[16][3];
    for (int h = 0; h < 16; ++h) {
        float a0 = 0, a1 = 0, a2 = 0;
        for (int i = 0; i < 3; ++i) { float w = wh[i * 16 + h]; a0 += v[i][0] * w; a1 += v[i][1] * w; a2 += v[i][2] * w; }
        vh[h][0] = a0; vh[h][1] = a1; vh[h][2] = a2;
        sv[6 + h] = sqrtf(fmaxf(a0 * a0 + a1 * a1 + a2 * a2, EPSV));
    }
    for (int o = 0; o < 128; ++o) {
        float acc = ws_b[o];
        for (int k = 0; k < 22; ++k) acc += sv[k] * ws_w[k * 128 + o];
        s_out[(size_t)nI * 128 + o] = acc;
    }
    for (int j = 0; j < 16; ++j) {
        float a0 = 0, a1 = 0, a2 = 0;
        for (int h = 0; h < 16; ++h) { float w = wv[h * 16 + j]; a0 += vh[h][0] * w; a1 += vh[h][1] * w; a2 += vh[h][2] * w; }
        v_out[(size_t)nI * 48 + j * 3 + 0] = a0;
        v_out[(size_t)nI * 48 + j * 3 + 1] = a1;
        v_out[(size_t)nI * 48 + j * 3 + 2] = a2;
    }
}

// ---------------------------------------------------------------------------
// W_e embedding: LN(32) + GVP(32,1 -> 32,1), no activations
// ---------------------------------------------------------------------------
__global__ void embed_edges_k(const float* __restrict__ s_in, const float* __restrict__ v_in,
                              const float* __restrict__ g, const float* __restrict__ be,
                              const float* __restrict__ wh,   /* 1x1 */
                              const float* __restrict__ ws_w, /* 33x32 */
                              const float* __restrict__ ws_b,
                              const float* __restrict__ wv,   /* 1x1 */
                              float* __restrict__ s_out, float* __restrict__ v_out)
{
    int e = blockIdx.x * blockDim.x + threadIdx.x;
    if (e >= NE) return;
    float s[33];
    for (int i = 0; i < 32; ++i) s[i] = s_in[(size_t)e * 32 + i];
    ln_inplace(s, 32, g, be);
    float v0 = v_in[(size_t)e * 3 + 0], v1 = v_in[(size_t)e * 3 + 1], v2 = v_in[(size_t)e * 3 + 2];
    float rms = sqrtf(fmaxf(v0 * v0 + v1 * v1 + v2 * v2, EPSV));   // vi==1 channel
    v0 /= rms; v1 /= rms; v2 /= rms;
    float w00 = wh[0];
    float h0 = v0 * w00, h1 = v1 * w00, h2 = v2 * w00;
    s[32] = sqrtf(fmaxf(h0 * h0 + h1 * h1 + h2 * h2, EPSV));
    for (int o = 0; o < 32; ++o) {
        float acc = ws_b[o];
        for (int k = 0; k < 33; ++k) acc += s[k] * ws_w[k * 32 + o];
        s_out[(size_t)e * 32 + o] = acc;
    }
    float wv0 = wv[0];
    v_out[(size_t)e * 3 + 0] = h0 * wv0;
    v_out[(size_t)e * 3 + 1] = h1 * wv0;
    v_out[(size_t)e * 3 + 2] = h2 * wv0;
}

// ---------------------------------------------------------------------------
// Build msg0 input: A[e] = bf16[ s_src(128) | e_s(32) | s_dst(128) | vn(33) | 0pad ]
// stride 352. Also vh = mv @ wh (33x33) stored bf16.
// ---------------------------------------------------------------------------
__global__ void msg_build0_k(const float* __restrict__ ns, const float* __restrict__ nv,
                             const float* __restrict__ es, const float* __restrict__ ev,
                             const int* __restrict__ ei,
                             const float* __restrict__ wh, /* 33x33 */
                             unsigned short* __restrict__ A, unsigned short* __restrict__ vhout)
{
    int e = blockIdx.x * blockDim.x + threadIdx.x;
    if (e >= NE) return;
    int src = ei[e], dst = ei[NE + e];
    unsigned short* Ar = A + (size_t)e * 352;
    for (int j = 0; j < 128; ++j) Ar[j]       = f2bf(ns[(size_t)src * 128 + j]);
    for (int j = 0; j < 32;  ++j) Ar[128 + j] = f2bf(es[(size_t)e * 32 + j]);
    for (int j = 0; j < 128; ++j) Ar[160 + j] = f2bf(ns[(size_t)dst * 128 + j]);
    float mv[33][3];
    for (int i = 0; i < 16; ++i)
        for (int c = 0; c < 3; ++c) mv[i][c] = nv[(size_t)src * 48 + i * 3 + c];
    for (int c = 0; c < 3; ++c) mv[16][c] = ev[(size_t)e * 3 + c];
    for (int i = 0; i < 16; ++i)
        for (int c = 0; c < 3; ++c) mv[17 + i][c] = nv[(size_t)dst * 48 + i * 3 + c];
    for (int h = 0; h < 33; ++h) {
        float a0 = 0, a1 = 0, a2 = 0;
        for (int i = 0; i < 33; ++i) { float w = wh[i * 33 + h]; a0 += mv[i][0] * w; a1 += mv[i][1] * w; a2 += mv[i][2] * w; }
        vhout[(size_t)e * 99 + h * 3 + 0] = f2bf(a0);
        vhout[(size_t)e * 99 + h * 3 + 1] = f2bf(a1);
        vhout[(size_t)e * 99 + h * 3 + 2] = f2bf(a2);
        Ar[288 + h] = f2bf(sqrtf(fmaxf(a0 * a0 + a1 * a1 + a2 * a2, EPSV)));
    }
    for (int j = 321; j < 352; ++j) Ar[j] = 0;
}

// ---------------------------------------------------------------------------
// After msgN gemm: vo = sigmoid-gated (vh @ wv), vh' = vo @ wh_next,
// A2[e] = bf16[ relu(C)(128) | vn'(16) | 0pad ] stride 160. H = prev hidden.
// ---------------------------------------------------------------------------
__global__ void msg_mid_k(const float* __restrict__ C, const unsigned short* __restrict__ vhprev,
                          int H, const float* __restrict__ wv, /* Hx16 */
                          const float* __restrict__ wh_next,   /* 16x16 */
                          unsigned short* __restrict__ A2, unsigned short* __restrict__ vhnext)
{
    int e = blockIdx.x * blockDim.x + threadIdx.x;
    if (e >= NE) return;
    float vh[33][3];
    for (int h = 0; h < H; ++h)
        for (int c = 0; c < 3; ++c) vh[h][c] = bf2f(vhprev[(size_t)e * (H * 3) + h * 3 + c]);
    float vo[16][3];
    for (int j = 0; j < 16; ++j) {
        float a0 = 0, a1 = 0, a2 = 0;
        for (int h = 0; h < H; ++h) { float w = wv[h * 16 + j]; a0 += vh[h][0] * w; a1 += vh[h][1] * w; a2 += vh[h][2] * w; }
        float gt = sigm(sqrtf(fmaxf(a0 * a0 + a1 * a1 + a2 * a2, EPSV)));
        vo[j][0] = a0 * gt; vo[j][1] = a1 * gt; vo[j][2] = a2 * gt;
    }
    unsigned short* Ar = A2 + (size_t)e * 160;
    for (int j = 0; j < 128; ++j) Ar[j] = f2bf(fmaxf(C[(size_t)e * 128 + j], 0.f));
    for (int h = 0; h < 16; ++h) {
        float a0 = 0, a1 = 0, a2 = 0;
        for (int j = 0; j < 16; ++j) { float w = wh_next[j * 16 + h]; a0 += vo[j][0] * w; a1 += vo[j][1] * w; a2 += vo[j][2] * w; }
        vhnext[(size_t)e * 48 + h * 3 + 0] = f2bf(a0);
        vhnext[(size_t)e * 48 + h * 3 + 1] = f2bf(a1);
        vhnext[(size_t)e * 48 + h * 3 + 2] = f2bf(a2);
        Ar[128 + h] = f2bf(sqrtf(fmaxf(a0 * a0 + a1 * a1 + a2 * a2, EPSV)));
    }
    for (int j = 144; j < 160; ++j) Ar[j] = 0;
}

// ---------------------------------------------------------------------------
// msg2 vector output (no gate) + scatter-add aggregation. One thread per
// (edge, feature); f in [0,177): 0..127 scalar, 128..175 vector, 176 count.
// ---------------------------------------------------------------------------
__global__ void msg_agg_k(const float* __restrict__ C, const unsigned short* __restrict__ vh2,
                          const float* __restrict__ wv, /* 16x16 */
                          const int* __restrict__ ei,
                          float* __restrict__ agg_s, float* __restrict__ agg_v,
                          float* __restrict__ cnt)
{
    long long t = (long long)blockIdx.x * blockDim.x + threadIdx.x;
    if (t >= (long long)NE * 177) return;
    int e = (int)(t / 177);
    int f = (int)(t - (long long)e * 177);
    int dst = ei[NE + e];
    if (f < 128) {
        atomicAdd(&agg_s[(size_t)dst * 128 + f], C[(size_t)e * 128 + f]);
    } else if (f < 176) {
        int rr = f - 128, j = rr / 3, c = rr - j * 3;
        float acc = 0.f;
        for (int h = 0; h < 16; ++h) acc += bf2f(vh2[(size_t)e * 48 + h * 3 + c]) * wv[h * 16 + j];
        atomicAdd(&agg_v[(size_t)dst * 48 + j * 3 + c], acc);
    } else {
        atomicAdd(&cnt[dst], 1.f);
    }
}

// ---------------------------------------------------------------------------
// residual + norm0, and build ff0 GEMM input A = [s(128)|vn(32)] stride 160
// ---------------------------------------------------------------------------
__global__ void node_update0_k(const float* __restrict__ ns_in, const float* __restrict__ nv_in,
                               const float* __restrict__ aggs, const float* __restrict__ aggv,
                               const float* __restrict__ cnt,
                               const float* __restrict__ g, const float* __restrict__ be,
                               const float* __restrict__ wh, /* 16x32 */
                               float* __restrict__ ns_out, float* __restrict__ nv_out,
                               unsigned short* __restrict__ A, unsigned short* __restrict__ vhout)
{
    int nI = blockIdx.x * blockDim.x + threadIdx.x;
    if (nI >= NN) return;
    float cn = fmaxf(cnt[nI], 1.f);
    float s[128];
    for (int j = 0; j < 128; ++j)
        s[j] = ns_in[(size_t)nI * 128 + j] + aggs[(size_t)nI * 128 + j] / cn;
    ln_inplace(s, 128, g, be);
    float v[16][3]; float q = 0.f;
    for (int i = 0; i < 16; ++i) {
        float ss = 0.f;
        for (int c = 0; c < 3; ++c) {
            v[i][c] = nv_in[(size_t)nI * 48 + i * 3 + c] + aggv[(size_t)nI * 48 + i * 3 + c] / cn;
            ss += v[i][c] * v[i][c];
        }
        q += fmaxf(ss, EPSV);
    }
    float rms = sqrtf(q / 16.f);
    for (int i = 0; i < 16; ++i) for (int c = 0; c < 3; ++c) v[i][c] /= rms;
    for (int j = 0; j < 128; ++j) ns_out[(size_t)nI * 128 + j] = s[j];
    for (int i = 0; i < 16; ++i)
        for (int c = 0; c < 3; ++c) nv_out[(size_t)nI * 48 + i * 3 + c] = v[i][c];
    unsigned short* Ar = A + (size_t)nI * 160;
    for (int j = 0; j < 128; ++j) Ar[j] = f2bf(s[j]);
    for (int h = 0; h < 32; ++h) {
        float a0 = 0, a1 = 0, a2 = 0;
        for (int i = 0; i < 16; ++i) { float w = wh[i * 32 + h]; a0 += v[i][0] * w; a1 += v[i][1] * w; a2 += v[i][2] * w; }
        vhout[(size_t)nI * 96 + h * 3 + 0] = f2bf(a0);
        vhout[(size_t)nI * 96 + h * 3 + 1] = f2bf(a1);
        vhout[(size_t)nI * 96 + h * 3 + 2] = f2bf(a2);
        Ar[128 + h] = f2bf(sqrtf(fmaxf(a0 * a0 + a1 * a1 + a2 * a2, EPSV)));
    }
}

// ---------------------------------------------------------------------------
// ff0 activations + build ff1 GEMM input A = [relu(Cff)(512)|vn1(32)] stride 544
// ---------------------------------------------------------------------------
__global__ void node_ff_mid_k(const float* __restrict__ Cff, const unsigned short* __restrict__ vhin,
                              const float* __restrict__ wv,  /* 32x32 */
                              const float* __restrict__ wh1, /* 32x32 */
                              unsigned short* __restrict__ A, unsigned short* __restrict__ vhout)
{
    int nI = blockIdx.x * blockDim.x + threadIdx.x;
    if (nI >= NN) return;
    float vh[32][3];
    for (int h = 0; h < 32; ++h)
        for (int c = 0; c < 3; ++c) vh[h][c] = bf2f(vhin[(size_t)nI * 96 + h * 3 + c]);
    float vo[32][3];
    for (int j = 0; j < 32; ++j) {
        float a0 = 0, a1 = 0, a2 = 0;
        for (int h = 0; h < 32; ++h) { float w = wv[h * 32 + j]; a0 += vh[h][0] * w; a1 += vh[h][1] * w; a2 += vh[h][2] * w; }
        float gt = sigm(sqrtf(fmaxf(a0 * a0 + a1 * a1 + a2 * a2, EPSV)));
        vo[j][0] = a0 * gt; vo[j][1] = a1 * gt; vo[j][2] = a2 * gt;
    }
    unsigned short* Ar = A + (size_t)nI * 544;
    for (int j = 0; j < 512; ++j) Ar[j] = f2bf(fmaxf(Cff[(size_t)nI * 512 + j], 0.f));
    for (int h = 0; h < 32; ++h) {
        float a0 = 0, a1 = 0, a2 = 0;
        for (int j = 0; j < 32; ++j) { float w = wh1[j * 32 + h]; a0 += vo[j][0] * w; a1 += vo[j][1] * w; a2 += vo[j][2] * w; }
        vhout[(size_t)nI * 96 + h * 3 + 0] = f2bf(a0);
        vhout[(size_t)nI * 96 + h * 3 + 1] = f2bf(a1);
        vhout[(size_t)nI * 96 + h * 3 + 2] = f2bf(a2);
        Ar[512 + h] = f2bf(sqrtf(fmaxf(a0 * a0 + a1 * a1 + a2 * a2, EPSV)));
    }
}

// ---------------------------------------------------------------------------
// ff residual + norm1
// ---------------------------------------------------------------------------
__global__ void node_update1_k(const float* __restrict__ Cff2, const unsigned short* __restrict__ vh1,
                               const float* __restrict__ wv, /* 32x16 */
                               const float* __restrict__ ns1, const float* __restrict__ nv1,
                               const float* __restrict__ g, const float* __restrict__ be,
                               float* __restrict__ ns_out, float* __restrict__ nv_out)
{
    int nI = blockIdx.x * blockDim.x + threadIdx.x;
    if (nI >= NN) return;
    float s[128];
    for (int j = 0; j < 128; ++j)
        s[j] = ns1[(size_t)nI * 128 + j] + Cff2[(size_t)nI * 128 + j];
    ln_inplace(s, 128, g, be);
    float v[16][3]; float q = 0.f;
    for (int j = 0; j < 16; ++j) {
        float ss = 0.f;
        for (int c = 0; c < 3; ++c) {
            float dv = 0.f;
            for (int h = 0; h < 32; ++h) dv += bf2f(vh1[(size_t)nI * 96 + h * 3 + c]) * wv[h * 16 + j];
            v[j][c] = nv1[(size_t)nI * 48 + j * 3 + c] + dv;
            ss += v[j][c] * v[j][c];
        }
        q += fmaxf(ss, EPSV);
    }
    float rms = sqrtf(q / 16.f);
    for (int j = 0; j < 128; ++j) ns_out[(size_t)nI * 128 + j] = s[j];
    for (int j = 0; j < 16; ++j)
        for (int c = 0; c < 3; ++c) nv_out[(size_t)nI * 48 + j * 3 + c] = v[j][c] / rms;
}

// ---------------------------------------------------------------------------
// W_out: LN + build A = [s(128)|vn(16)|0(16)] stride 160 (GEMM+relu -> d_out)
// ---------------------------------------------------------------------------
__global__ void out_build_k(const float* __restrict__ ns_in, const float* __restrict__ nv_in,
                            const float* __restrict__ g, const float* __restrict__ be,
                            const float* __restrict__ wh, /* 16x16 */
                            unsigned short* __restrict__ A)
{
    int nI = blockIdx.x * blockDim.x + threadIdx.x;
    if (nI >= NN) return;
    float s[128];
    for (int j = 0; j < 128; ++j) s[j] = ns_in[(size_t)nI * 128 + j];
    ln_inplace(s, 128, g, be);
    float v[16][3]; float q = 0.f;
    for (int i = 0; i < 16; ++i) {
        float ss = 0.f;
        for (int c = 0; c < 3; ++c) { v[i][c] = nv_in[(size_t)nI * 48 + i * 3 + c]; ss += v[i][c] * v[i][c]; }
        q += fmaxf(ss, EPSV);
    }
    float rms = sqrtf(q / 16.f);
    for (int i = 0; i < 16; ++i) for (int c = 0; c < 3; ++c) v[i][c] /= rms;
    unsigned short* Ar = A + (size_t)nI * 160;
    for (int j = 0; j < 128; ++j) Ar[j] = f2bf(s[j]);
    for (int h = 0; h < 16; ++h) {
        float a0 = 0, a1 = 0, a2 = 0;
        for (int i = 0; i < 16; ++i) { float w = wh[i * 16 + h]; a0 += v[i][0] * w; a1 += v[i][1] * w; a2 += v[i][2] * w; }
        Ar[128 + h] = f2bf(sqrtf(fmaxf(a0 * a0 + a1 * a1 + a2 * a2, EPSV)));
    }
    for (int j = 144; j < 160; ++j) Ar[j] = 0;
}

// ===========================================================================
extern "C" void kernel_launch(void* const* d_in, const int* in_sizes, int n_in,
                              void* d_out, int out_size, void* d_ws, size_t ws_size,
                              hipStream_t stream)
{
    (void)in_sizes; (void)n_in; (void)out_size; (void)ws_size;
    // jax tree_flatten (alphabetical) leaf order:
    // 0 batch, 1 edge_index, 2 edge_s, 3 edge_v, 4 node_s, 5 node_v, then params:
    // We(gvp: wh,ws_b,ws_w,wv; ln: beta,gamma) 6..11
    // Wout(gvp: wh,ws_b,ws_w; ln: beta,gamma)  12..16
    // Wv (gvp: wh,ws_b,ws_w,wv; ln: beta,gamma) 17..22
    // layers[L] base 23+24L: ff0(wh,ws_b,ws_w,wv) ff1(..) msg0(..) msg1(..) msg2(..)
    //                        norm0(beta,gamma) norm1(beta,gamma)
    const int*   ei     = (const int*)d_in[1];
    const float* edge_s = (const float*)d_in[2];
    const float* edge_v = (const float*)d_in[3];
    const float* node_s = (const float*)d_in[4];
    const float* node_v = (const float*)d_in[5];
    auto P  = [&](int i) { return (const float*)d_in[i]; };
    auto LP = [&](int L, int o) { return (const float*)d_in[23 + 24 * L + o]; };

    // allow up to 176 KB dynamic LDS for the GEMM (ff0 weights: 512 x 168 bf16)
    hipFuncSetAttribute((const void*)gemm_bf16_k,
                        hipFuncAttributeMaxDynamicSharedMemorySize, 176 * 1024);

    // ---- carve workspace (~335 MB) ----
    char* wp = (char*)d_ws;
    auto carve = [&](size_t bytes) -> void* {
        void* p = (void*)wp; wp += (bytes + 255) & ~(size_t)255; return p;
    };
    float* ns   = (float*)carve((size_t)NN * 128 * 4);
    float* nv   = (float*)carve((size_t)NN * 48 * 4);
    float* ns1  = (float*)carve((size_t)NN * 128 * 4);
    float* nv1  = (float*)carve((size_t)NN * 48 * 4);
    float* esb  = (float*)carve((size_t)NE * 32 * 4);
    float* evb  = (float*)carve((size_t)NE * 3 * 4);
    unsigned short* Ae  = (unsigned short*)carve((size_t)NE * 352 * 2);
    unsigned short* vhA = (unsigned short*)carve((size_t)NE * 99 * 2);
    unsigned short* vhB = (unsigned short*)carve((size_t)NE * 48 * 2);
    float* Ce   = (float*)carve((size_t)NE * 128 * 4);
    float* aggs = (float*)carve((size_t)NN * 128 * 4);
    float* aggv = (float*)carve((size_t)NN * 48 * 4);
    float* cntb = (float*)carve((size_t)NN * 4);
    unsigned short* An = (unsigned short*)carve((size_t)NN * 544 * 2);
    float* Cff  = (float*)carve((size_t)NN * 512 * 4);
    float* Cff2 = (float*)carve((size_t)NN * 128 * 4);
    unsigned short* vhF  = (unsigned short*)carve((size_t)NN * 96 * 2);
    unsigned short* vhF1 = (unsigned short*)carve((size_t)NN * 96 * 2);
    unsigned short *W0T[3], *W1T[3], *W2T[3], *F0T[3], *F1T[3];
    for (int L = 0; L < 3; ++L) {
        W0T[L] = (unsigned short*)carve((size_t)128 * 352 * 2);
        W1T[L] = (unsigned short*)carve((size_t)128 * 160 * 2);
        W2T[L] = (unsigned short*)carve((size_t)128 * 160 * 2);
        F0T[L] = (unsigned short*)carve((size_t)512 * 160 * 2);
        F1T[L] = (unsigned short*)carve((size_t)128 * 544 * 2);
    }
    unsigned short* WoT = (unsigned short*)carve((size_t)128 * 160 * 2);

    const int TB = 256;                      // 8 waves per block
    const int NB = (NN + TB - 1) / TB;
    const int EB = (NE + TB - 1) / TB;
    auto wconv = [&](const float* W, unsigned short* WT, int K, int N, int Kpad) {
        int t = N * Kpad;
        wconv_k<<<(t + TB - 1) / TB, TB, 0, stream>>>(W, WT, K, N, Kpad);
    };
    auto gemm = [&](const unsigned short* A, int lda, const unsigned short* BT,
                    const float* bias, float* C, int M, int N, int K, int act) {
        int strips = (M / 16) * (N / 128);       // 16x128 strips
        int blocks = (strips + 15) / 16;         // ~2 strips per wave
        if (blocks < 1) blocks = 1;
        size_t shbytes = (size_t)N * (K + 8) * 2;  // LDS-staged weights (padded)
        gemm_bf16_k<<<blocks, TB, shbytes, stream>>>(A, lda, BT, bias, C, M, N, K, act);
    };

    // ---- weight conversion (fp32 -> bf16 transposed, K padded to mult of 32) ----
    for (int L = 0; L < 3; ++L) {
        wconv(LP(L, 10), W0T[L], 321, 128, 352);   // msg0.ws_w
        wconv(LP(L, 14), W1T[L], 144, 128, 160);   // msg1.ws_w
        wconv(LP(L, 18), W2T[L], 144, 128, 160);   // msg2.ws_w
        wconv(LP(L, 2),  F0T[L], 160, 512, 160);   // ff0.ws_w
        wconv(LP(L, 6),  F1T[L], 544, 128, 544);   // ff1.ws_w
    }
    wconv(P(14), WoT, 144, 128, 160);              // Wout.ws_w

    // ---- embeddings ----
    embed_nodes_k<<<NB, TB, 0, stream>>>(node_s, node_v, P(22), P(21),
                                         P(17), P(19), P(18), P(20), ns, nv);
    embed_edges_k<<<EB, TB, 0, stream>>>(edge_s, edge_v, P(11), P(10),
                                         P(6), P(8), P(7), P(9), esb, evb);

    // ---- 3 GVPConv layers ----
    long long aggT = (long long)NE * 177;
    int aggB = (int)((aggT + TB - 1) / TB);
    for (int L = 0; L < 3; ++L) {
        zero_k<<<(NN * 128 + TB - 1) / TB, TB, 0, stream>>>(aggs, NN * 128);
        zero_k<<<(NN * 48 + TB - 1) / TB, TB, 0, stream>>>(aggv, NN * 48);
        zero_k<<<(NN + TB - 1) / TB, TB, 0, stream>>>(cntb, NN);

        msg_build0_k<<<EB, TB, 0, stream>>>(ns, nv, esb, evb, ei, LP(L, 8), Ae, vhA);
        gemm(Ae, 352, W0T[L], LP(L, 9),  Ce, NE, 128, 352, 0);
        msg_mid_k<<<EB, TB, 0, stream>>>(Ce, vhA, 33, LP(L, 11), LP(L, 12), Ae, vhB);
        gemm(Ae, 160, W1T[L], LP(L, 13), Ce, NE, 128, 160, 0);
        msg_mid_k<<<EB, TB, 0, stream>>>(Ce, vhB, 16, LP(L, 15), LP(L, 16), Ae, vhA);
        gemm(Ae, 160, W2T[L], LP(L, 17), Ce, NE, 128, 160, 0);
        msg_agg_k<<<aggB, TB, 0, stream>>>(Ce, vhA, LP(L, 19), ei, aggs, aggv, cntb);

        node_update0_k<<<NB, TB, 0, stream>>>(ns, nv, aggs, aggv, cntb,
                                              LP(L, 21), LP(L, 20), LP(L, 0),
                                              ns1, nv1, An, vhF);
        gemm(An, 160, F0T[L], LP(L, 1), Cff, NN, 512, 160, 0);
        node_ff_mid_k<<<NB, TB, 0, stream>>>(Cff, vhF, LP(L, 3), LP(L, 4), An, vhF1);
        gemm(An, 544, F1T[L], LP(L, 5), Cff2, NN, 128, 544, 0);
        node_update1_k<<<NB, TB, 0, stream>>>(Cff2, vhF1, LP(L, 7), ns1, nv1,
                                              LP(L, 23), LP(L, 22), ns, nv);
    }

    // ---- W_out head: LN + GEMM + relu straight into d_out (pyg_split == identity) ----
    out_build_k<<<NB, TB, 0, stream>>>(ns, nv, P(16), P(15), P(12), An);
    gemm(An, 160, WoT, P(13), (float*)d_out, NN, 128, 160, 1);
}
